// MultiHeadAttention_46308337386103
// MI455X (gfx1250) — compile-verified, matrix operand order checked by reading, not run on hardware
//
#include <hip/hip_runtime.h>
#include <math.h>

#define B_ 4
#define L_ 1024
#define D_ 1024
#define H_ 16

typedef unsigned short u16;
typedef unsigned int   u32;
typedef __attribute__((ext_vector_type(16))) __bf16 v16bf;
typedef __attribute__((ext_vector_type(8)))  float  v8f;
typedef __attribute__((ext_vector_type(4)))  unsigned int u32x4;
typedef __attribute__((ext_vector_type(2)))  unsigned int u32x2;
typedef __attribute__((ext_vector_type(4)))  float  f32x4;

union FragU { v16bf v; u32x4 q[2]; };

__device__ inline u16 f2bf(float f) {
  u32 u = __builtin_bit_cast(u32, f);
  u32 r = u + 0x7fffu + ((u >> 16) & 1u);   // round to nearest even
  return (u16)(r >> 16);
}

__device__ inline v8f zero8() { v8f z = {0.f,0.f,0.f,0.f,0.f,0.f,0.f,0.f}; return z; }

__device__ inline v8f wmma_bf16(v16bf a, v16bf b, v8f c) {
  return __builtin_amdgcn_wmma_f32_16x16x32_bf16(false, a, false, b, (short)0, c, false, false);
}

// Load a 16x32 bf16 fragment (A layout) from a row-major matrix.
// For B fragments, pass the "weight style" matrix W[N][K] (B = W^T): identical pattern.
__device__ inline v16bf load_frag(const u16* base, int ld, int row0, int k0, int lane) {
  int m  = lane & 15;
  int kb = (lane >> 4) << 3;
  const u16* p = base + (size_t)(row0 + m) * ld + (k0 + kb);
  FragU f;
  f.q[0] = *reinterpret_cast<const u32x4*>(p);
  f.q[1] = *reinterpret_cast<const u32x4*>(p + 16);
  return f.v;
}

// ---- f32 -> bf16, 4 elements per thread --------------------------------
__global__ __launch_bounds__(256)
void cvt_bf16_kernel(const float* __restrict__ in, u16* __restrict__ out, int n4) {
  int i = blockIdx.x * 256 + threadIdx.x;
  if (i < n4) {
    f32x4 xv = reinterpret_cast<const f32x4*>(in)[i];
    u32x2 o;
    o.x = (u32)f2bf(xv.x) | ((u32)f2bf(xv.y) << 16);
    o.y = (u32)f2bf(xv.z) | ((u32)f2bf(xv.w) << 16);
    reinterpret_cast<u32x2*>(out)[i] = o;
  }
}

// ---- projection GEMM: 32x64 per wave, unroll-by-2 ping-pong -------------
// out_bf16 = act[4096,1024] @ W[1024,1024]^T + bias, scattered head-major.
// transpose_v==0 -> [B,H,L,64]; transpose_v==1 -> [B,H,64,L]
__global__ __launch_bounds__(256)
void proj_kernel(const u16* __restrict__ act, const u16* __restrict__ w,
                 const float* __restrict__ bias, u16* __restrict__ out, int transpose_v) {
  int nt = blockIdx.x & 15;          // 16 n-tiles of 64
  int mt = blockIdx.x >> 4;          // 16 m-tiles of 256
  int wave = threadIdx.x >> 5, lane = threadIdx.x & 31;
  int mbase = mt * 256 + wave * 32;
  int nbase = nt * 64;
  v8f acc[8] = {zero8(), zero8(), zero8(), zero8(), zero8(), zero8(), zero8(), zero8()};

  v16bf a0A = load_frag(act, 1024, mbase,      0, lane);
  v16bf a1A = load_frag(act, 1024, mbase + 16, 0, lane);
  v16bf bA[4];
#pragma unroll
  for (int j = 0; j < 4; ++j) bA[j] = load_frag(w, 1024, nbase + j * 16, 0, lane);
  v16bf a0B = load_frag(act, 1024, mbase,      32, lane);
  v16bf a1B = load_frag(act, 1024, mbase + 16, 32, lane);
  v16bf bB[4];
#pragma unroll
  for (int j = 0; j < 4; ++j) bB[j] = load_frag(w, 1024, nbase + j * 16, 32, lane);

  for (int k0 = 0; k0 < 1024; k0 += 64) {
    // consume buffer A (k0), then refill it for k0+64
#pragma unroll
    for (int j = 0; j < 4; ++j) {
      acc[j * 2 + 0] = wmma_bf16(a0A, bA[j], acc[j * 2 + 0]);
      acc[j * 2 + 1] = wmma_bf16(a1A, bA[j], acc[j * 2 + 1]);
    }
    int ka = (k0 + 64) & 1023;                  // wrapped prefetch (branch-free)
    a0A = load_frag(act, 1024, mbase,      ka, lane);
    a1A = load_frag(act, 1024, mbase + 16, ka, lane);
#pragma unroll
    for (int j = 0; j < 4; ++j) bA[j] = load_frag(w, 1024, nbase + j * 16, ka, lane);
    // consume buffer B (k0+32), then refill it for k0+96
#pragma unroll
    for (int j = 0; j < 4; ++j) {
      acc[j * 2 + 0] = wmma_bf16(a0B, bB[j], acc[j * 2 + 0]);
      acc[j * 2 + 1] = wmma_bf16(a1B, bB[j], acc[j * 2 + 1]);
    }
    int kb2 = (k0 + 96) & 1023;
    a0B = load_frag(act, 1024, mbase,      kb2, lane);
    a1B = load_frag(act, 1024, mbase + 16, kb2, lane);
#pragma unroll
    for (int j = 0; j < 4; ++j) bB[j] = load_frag(w, 1024, nbase + j * 16, kb2, lane);
  }

  int hi = lane >> 4, nloc = lane & 15;
#pragma unroll
  for (int j = 0; j < 4; ++j) {
    int n = nbase + j * 16 + nloc;
    int h = n >> 6, d = n & 63;
    float bs = bias[n];
#pragma unroll
    for (int half = 0; half < 2; ++half) {
#pragma unroll
      for (int r = 0; r < 8; ++r) {
        int m = mbase + half * 16 + r + hi * 8;
        int b = m >> 10, l = m & 1023;
        float val = acc[j * 2 + half][r] + bs;
        size_t idx = transpose_v
            ? ((size_t)((b * H_ + h) * 64 + d) * L_ + l)
            : ((size_t)((b * H_ + h) * L_ + l) * 64 + d);
        out[idx] = f2bf(val);
      }
    }
  }
}

// ---- fused attention: 16 q-rows per wave, two-pass online softmax -------
__global__ __launch_bounds__(256)
void attn_kernel(const u16* __restrict__ qh, const u16* __restrict__ kh,
                 const u16* __restrict__ vt, const int* __restrict__ seq_len,
                 float* __restrict__ attn_out, u16* __restrict__ obuf) {
  int qt = blockIdx.x & 7;           // 8 q-tiles of 128
  int bh = blockIdx.x >> 3;          // 0..63
  int b = bh >> 4, h = bh & 15;
  int wave = threadIdx.x >> 5, lane = threadIdx.x & 31;
  int qbase = qt * 128 + wave * 16;
  int slen = seq_len[b];
  const u16* Q = qh + (size_t)bh * L_ * 64;
  const u16* K = kh + (size_t)bh * L_ * 64;
  const u16* V = vt + (size_t)bh * 64 * L_;   // [64, L]
  float* P = attn_out + ((size_t)(h * B_ + b) * L_ + qbase) * L_;

  __shared__ alignas(16) u16 lds_p[8][16 * 32];
  u16* lp = &lds_p[wave][0];

  v16bf a0 = load_frag(Q, 64, qbase, 0, lane);
  v16bf a1 = load_frag(Q, 64, qbase, 32, lane);

  int hi = lane >> 4, nloc = lane & 15;
  const float scale = 0.125f;                 // 1/sqrt(64)
  const float NEG_INF = -__builtin_inff();

  float M[8], S[8];
#pragma unroll
  for (int r = 0; r < 8; ++r) { M[r] = NEG_INF; S[r] = 0.f; }

  // ---- pass 1: per-lane online (max,sum), ping-pong K buffers
  v16bf k0A = load_frag(K, 64,  0, 0, lane);
  v16bf k1A = load_frag(K, 64,  0, 32, lane);
  v16bf k0B = load_frag(K, 64, 16, 0, lane);
  v16bf k1B = load_frag(K, 64, 16, 32, lane);
  for (int kt = 0; kt < L_; kt += 32) {
    v8f c = zero8();
    c = wmma_bf16(a0, k0A, c);
    c = wmma_bf16(a1, k1A, c);
    int ka = (kt + 32) & (L_ - 1);
    k0A = load_frag(K, 64, ka, 0, lane);
    k1A = load_frag(K, 64, ka, 32, lane);
    if ((kt + nloc) < slen) {
#pragma unroll
      for (int r = 0; r < 8; ++r) {
        float s = c[r] * scale;
        float Mn = fmaxf(M[r], s);
        S[r] = S[r] * __expf(M[r] - Mn) + __expf(s - Mn);
        M[r] = Mn;
      }
    }
    v8f c2 = zero8();
    c2 = wmma_bf16(a0, k0B, c2);
    c2 = wmma_bf16(a1, k1B, c2);
    int kb2 = (kt + 48) & (L_ - 1);
    k0B = load_frag(K, 64, kb2, 0, lane);
    k1B = load_frag(K, 64, kb2, 32, lane);
    if ((kt + 16 + nloc) < slen) {
#pragma unroll
      for (int r = 0; r < 8; ++r) {
        float s = c2[r] * scale;
        float Mn = fmaxf(M[r], s);
        S[r] = S[r] * __expf(M[r] - Mn) + __expf(s - Mn);
        M[r] = Mn;
      }
    }
  }
  // merge (max,sum) across the 16-lane group sharing each row
#pragma unroll
  for (int r = 0; r < 8; ++r) {
#pragma unroll
    for (int off = 1; off < 16; off <<= 1) {
      float Mo = __shfl_xor(M[r], off, 32);
      float So = __shfl_xor(S[r], off, 32);
      float Mn = fmaxf(M[r], Mo);
      float w1 = (M[r] == NEG_INF) ? 0.f : __expf(M[r] - Mn);
      float w2 = (Mo   == NEG_INF) ? 0.f : __expf(Mo   - Mn);
      S[r] = S[r] * w1 + So * w2;
      M[r] = Mn;
    }
  }
  float fnorm[8];
#pragma unroll
  for (int r = 0; r < 8; ++r) {
    int qrow = qbase + r + hi * 8;
    fnorm[r] = (qrow < slen && S[r] > 0.f) ? 1.f / S[r] : 0.f;  // query mask folded in
  }

  // ---- pass 2: recompute scores, emit probs (NT stores), accumulate P*V
  // LDS tile is wave-private; DS ops from one wave execute in order (DScnt),
  // so only a compiler reordering fence is needed around the LDS round-trip.
  v8f oacc[4] = {zero8(), zero8(), zero8(), zero8()};
  k0A = load_frag(K, 64,  0, 0, lane);
  k1A = load_frag(K, 64,  0, 32, lane);
  k0B = load_frag(K, 64, 16, 0, lane);
  k1B = load_frag(K, 64, 16, 32, lane);
  for (int k2 = 0; k2 < L_; k2 += 32) {
    v16bf vfrag[4];
#pragma unroll
    for (int j = 0; j < 4; ++j) vfrag[j] = load_frag(V, L_, j * 16, k2, lane);  // early issue

    // subtile 0 (cols k2..k2+15)
    {
      v8f c = zero8();
      c = wmma_bf16(a0, k0A, c);
      c = wmma_bf16(a1, k1A, c);
      int ka = (k2 + 32) & (L_ - 1);
      k0A = load_frag(K, 64, ka, 0, lane);
      k1A = load_frag(K, 64, ka, 32, lane);
      bool kvalid = (k2 + nloc) < slen;
#pragma unroll
      for (int r = 0; r < 8; ++r) {
        float p = kvalid ? __expf(c[r] * scale - M[r]) * fnorm[r] : 0.f;
        int mrow = r + hi * 8;
        __builtin_nontemporal_store(p, &P[(size_t)mrow * L_ + k2 + nloc]);   // 256MB stream
        lp[mrow * 32 + nloc] = f2bf(p);                                      // C-layout -> LDS
      }
    }
    // subtile 1 (cols k2+16..k2+31)
    {
      v8f c = zero8();
      c = wmma_bf16(a0, k0B, c);
      c = wmma_bf16(a1, k1B, c);
      int kb2 = (k2 + 48) & (L_ - 1);
      k0B = load_frag(K, 64, kb2, 0, lane);
      k1B = load_frag(K, 64, kb2, 32, lane);
      bool kvalid = (k2 + 16 + nloc) < slen;
#pragma unroll
      for (int r = 0; r < 8; ++r) {
        float p = kvalid ? __expf(c[r] * scale - M[r]) * fnorm[r] : 0.f;
        int mrow = r + hi * 8;
        __builtin_nontemporal_store(p, &P[(size_t)mrow * L_ + k2 + 16 + nloc]);
        lp[mrow * 32 + 16 + nloc] = f2bf(p);
      }
    }
    asm volatile("" ::: "memory");   // keep ds_loads after ds_stores; HW DS is in-order
    {
      // LDS 16x32 tile -> A fragment
      int m = lane & 15;
      int kb = hi << 3;
      const u16* pp = lp + m * 32 + kb;
      FragU fa;
      fa.q[0] = *reinterpret_cast<const u32x4*>(pp);
      fa.q[1] = *reinterpret_cast<const u32x4*>(pp + 16);
#pragma unroll
      for (int j = 0; j < 4; ++j)
        oacc[j] = wmma_bf16(fa.v, vfrag[j], oacc[j]);
    }
    asm volatile("" ::: "memory");   // next iter's stores stay after these loads
  }

  // store O tile (16 x 64) bf16 into obuf[4096,1024] at columns h*64..h*64+63
#pragma unroll
  for (int j = 0; j < 4; ++j) {
#pragma unroll
    for (int r = 0; r < 8; ++r) {
      int l = qbase + r + hi * 8;
      int col = h * 64 + j * 16 + nloc;
      obuf[(size_t)(b * L_ + l) * 1024 + col] = f2bf(oacc[j][r]);
    }
  }
}

// ---- x = O @ fc_w^T + fc_b + residual (f32), 32x64/wave, ping-pong ------
__global__ __launch_bounds__(256)
void fc_kernel(const u16* __restrict__ act, const u16* __restrict__ w,
               const float* __restrict__ bias, const float* __restrict__ resid,
               float* __restrict__ xout) {
  int nt = blockIdx.x & 15;
  int mt = blockIdx.x >> 4;
  int wave = threadIdx.x >> 5, lane = threadIdx.x & 31;
  int mbase = mt * 256 + wave * 32;
  int nbase = nt * 64;
  v8f acc[8] = {zero8(), zero8(), zero8(), zero8(), zero8(), zero8(), zero8(), zero8()};

  v16bf a0A = load_frag(act, 1024, mbase,      0, lane);
  v16bf a1A = load_frag(act, 1024, mbase + 16, 0, lane);
  v16bf bA[4];
#pragma unroll
  for (int j = 0; j < 4; ++j) bA[j] = load_frag(w, 1024, nbase + j * 16, 0, lane);
  v16bf a0B = load_frag(act, 1024, mbase,      32, lane);
  v16bf a1B = load_frag(act, 1024, mbase + 16, 32, lane);
  v16bf bB[4];
#pragma unroll
  for (int j = 0; j < 4; ++j) bB[j] = load_frag(w, 1024, nbase + j * 16, 32, lane);

  for (int k0 = 0; k0 < 1024; k0 += 64) {
#pragma unroll
    for (int j = 0; j < 4; ++j) {
      acc[j * 2 + 0] = wmma_bf16(a0A, bA[j], acc[j * 2 + 0]);
      acc[j * 2 + 1] = wmma_bf16(a1A, bA[j], acc[j * 2 + 1]);
    }
    int ka = (k0 + 64) & 1023;
    a0A = load_frag(act, 1024, mbase,      ka, lane);
    a1A = load_frag(act, 1024, mbase + 16, ka, lane);
#pragma unroll
    for (int j = 0; j < 4; ++j) bA[j] = load_frag(w, 1024, nbase + j * 16, ka, lane);
#pragma unroll
    for (int j = 0; j < 4; ++j) {
      acc[j * 2 + 0] = wmma_bf16(a0B, bB[j], acc[j * 2 + 0]);
      acc[j * 2 + 1] = wmma_bf16(a1B, bB[j], acc[j * 2 + 1]);
    }
    int kb2 = (k0 + 96) & 1023;
    a0B = load_frag(act, 1024, mbase,      kb2, lane);
    a1B = load_frag(act, 1024, mbase + 16, kb2, lane);
#pragma unroll
    for (int j = 0; j < 4; ++j) bB[j] = load_frag(w, 1024, nbase + j * 16, kb2, lane);
  }

  int hi = lane >> 4, nloc = lane & 15;
#pragma unroll
  for (int j = 0; j < 4; ++j) {
    int n = nbase + j * 16 + nloc;
    float bs = bias[n];
#pragma unroll
    for (int half = 0; half < 2; ++half) {
#pragma unroll
      for (int r = 0; r < 8; ++r) {
        int m = mbase + half * 16 + r + hi * 8;
        size_t idx = (size_t)m * 1024 + n;
        xout[idx] = acc[j * 2 + half][r] + bs + resid[idx];
      }
    }
  }
}

// ---- LayerNorm over D=1024, one wave per row ----------------------------
__global__ __launch_bounds__(256)
void ln_kernel(const float* __restrict__ x, const float* __restrict__ g,
               const float* __restrict__ bb, float* __restrict__ out) {
  int wave = threadIdx.x >> 5, lane = threadIdx.x & 31;
  int row = blockIdx.x * 8 + wave;
  const float* xr = x + (size_t)row * 1024;
  float s = 0.f, s2 = 0.f;
#pragma unroll
  for (int i = 0; i < 32; ++i) {
    float v = xr[lane + i * 32];
    s += v; s2 += v * v;
  }
#pragma unroll
  for (int off = 16; off >= 1; off >>= 1) {
    s  += __shfl_xor(s,  off, 32);
    s2 += __shfl_xor(s2, off, 32);
  }
  float mu  = s  * (1.f / 1024.f);
  float var = s2 * (1.f / 1024.f) - mu * mu;
  float rinv = rsqrtf(var + 1e-5f);
  float* orow = out + (size_t)row * 1024;
#pragma unroll
  for (int i = 0; i < 32; ++i) {
    int c = lane + i * 32;
    float o = (xr[c] - mu) * rinv * g[c] + bb[c];
    __builtin_nontemporal_store(o, &orow[c]);
  }
}

extern "C" void kernel_launch(void* const* d_in, const int* in_sizes, int n_in,
                              void* d_out, int out_size, void* d_ws, size_t ws_size,
                              hipStream_t stream) {
  (void)in_sizes; (void)n_in; (void)out_size; (void)ws_size;
  const float* q    = (const float*)d_in[0];
  const float* k    = (const float*)d_in[1];
  const float* v    = (const float*)d_in[2];
  const int*   slen = (const int*)  d_in[3];
  const float* w_qs = (const float*)d_in[4];
  const float* b_qs = (const float*)d_in[5];
  const float* w_ks = (const float*)d_in[6];
  const float* b_ks = (const float*)d_in[7];
  const float* w_vs = (const float*)d_in[8];
  const float* b_vs = (const float*)d_in[9];
  const float* fc_w = (const float*)d_in[10];
  const float* fc_b = (const float*)d_in[11];
  const float* ln_g = (const float*)d_in[12];
  const float* ln_b = (const float*)d_in[13];

  float* out      = (float*)d_out;
  float* attn_out = out + (size_t)B_ * L_ * D_;

  const size_t NA = (size_t)B_ * L_ * D_;   // 4M elems
  const size_t NW = (size_t)D_ * D_;        // 1M elems
  u16* qb   = (u16*)d_ws;
  u16* kb   = qb  + NA;
  u16* vb   = kb  + NA;
  u16* wqb  = vb  + NA;
  u16* wkb  = wqb + NW;
  u16* wvb  = wkb + NW;
  u16* fwb  = wvb + NW;
  u16* qhb  = fwb + NW;
  u16* khb  = qhb + NA;
  u16* vtb  = khb + NA;
  u16* obuf = vtb + NA;
  float* xbuf = (float*)(obuf + NA);

  dim3 blk(256);
  cvt_bf16_kernel<<<dim3((unsigned)(NA / 1024)), blk, 0, stream>>>(q, qb, (int)(NA / 4));
  cvt_bf16_kernel<<<dim3((unsigned)(NA / 1024)), blk, 0, stream>>>(k, kb, (int)(NA / 4));
  cvt_bf16_kernel<<<dim3((unsigned)(NA / 1024)), blk, 0, stream>>>(v, vb, (int)(NA / 4));
  cvt_bf16_kernel<<<dim3((unsigned)(NW / 1024)), blk, 0, stream>>>(w_qs, wqb, (int)(NW / 4));
  cvt_bf16_kernel<<<dim3((unsigned)(NW / 1024)), blk, 0, stream>>>(w_ks, wkb, (int)(NW / 4));
  cvt_bf16_kernel<<<dim3((unsigned)(NW / 1024)), blk, 0, stream>>>(w_vs, wvb, (int)(NW / 4));
  cvt_bf16_kernel<<<dim3((unsigned)(NW / 1024)), blk, 0, stream>>>(fc_w, fwb, (int)(NW / 4));

  proj_kernel<<<dim3(256), blk, 0, stream>>>(qb, wqb, b_qs, qhb, 0);
  proj_kernel<<<dim3(256), blk, 0, stream>>>(kb, wkb, b_ks, khb, 0);
  proj_kernel<<<dim3(256), blk, 0, stream>>>(vb, wvb, b_vs, vtb, 1);

  attn_kernel<<<dim3(512), blk, 0, stream>>>(qhb, khb, vtb, slen, attn_out, obuf);

  fc_kernel<<<dim3(256), blk, 0, stream>>>(obuf, fwb, fc_b, q, xbuf);
  ln_kernel<<<dim3(512), blk, 0, stream>>>(xbuf, ln_g, ln_b, out);
}
